// TrafficGNN_53755810677034
// MI455X (gfx1250) — matrix operand order repaired and checked by reading, not compiled, not run on hardware
//
#include <hip/hip_runtime.h>
#include <hip/hip_bf16.h>

// ---------------------------------------------------------------------------
// TrafficGNN on MI455X (gfx1250): 2x GCN + GAT + linear head.
// GEMMs: bf16 WMMA (v_wmma_f32_16x16x32_bf16), f32 accumulate.
// W staging into LDS via Tensor Data Mover (tensor_load_to_lds + TENSORcnt).
// Aggregations: wave-per-edge float4 gather + global f32 atomic scatter
// (h matrices are 51.2MB -> L2 resident on the 192MB L2).
// ---------------------------------------------------------------------------

typedef __bf16 bf16_t;
typedef __attribute__((ext_vector_type(16))) __bf16 v16bf;
typedef __attribute__((ext_vector_type(8)))  float  v8f;
typedef __attribute__((ext_vector_type(4)))  unsigned int u32x4;
typedef __attribute__((ext_vector_type(8)))  int  i32x8;
typedef __attribute__((ext_vector_type(4)))  int  i32x4;

#define HDIM 128
#define CDIM 16

// ---------------- small utility kernels ----------------

__global__ __launch_bounds__(256) void k_fill_f32(float* __restrict__ p, long n, float v) {
  long i = (long)blockIdx.x * blockDim.x + threadIdx.x;
  if (i < n) p[i] = v;
}

__global__ __launch_bounds__(256) void k_deg(const int* __restrict__ ei,
                                             const float* __restrict__ ew,
                                             float* __restrict__ deg, int E) {
  int e = blockIdx.x * blockDim.x + threadIdx.x;
  if (e < E) atomicAdd(&deg[ei[E + e]], ew[e]);   // dst = ei[E+e]
}

__global__ __launch_bounds__(256) void k_dinv(const float* __restrict__ deg,
                                              float* __restrict__ dinv, int n) {
  int i = blockIdx.x * blockDim.x + threadIdx.x;
  if (i < n) { float d = deg[i]; dinv[i] = d > 0.f ? rsqrtf(d) : 0.f; }
}

__global__ __launch_bounds__(256) void k_f32_to_bf(const float* __restrict__ x,
                                                   bf16_t* __restrict__ o, long n) {
  long i = (long)blockIdx.x * blockDim.x + threadIdx.x;
  if (i < n) o[i] = (bf16_t)x[i];
}

// transpose + convert: o[n*K + k] = W[k*Nc + n]  (W is [K x Nc] row-major)
__global__ __launch_bounds__(256) void k_wT_to_bf(const float* __restrict__ W,
                                                  bf16_t* __restrict__ o, int K, int Nc) {
  int i = blockIdx.x * blockDim.x + threadIdx.x;
  if (i < K * Nc) {
    int k = i / Nc, nn = i % Nc;
    o[(size_t)nn * K + k] = (bf16_t)W[i];
  }
}

// ---------------- TDM staging: 128x128 bf16 linear tile -> LDS ----------------
// D# per CDNA5 ISA: group0 {count=1, lds_addr,
// global_addr, type=2}; group1 {data_size=2B, tensor/tile dims 128, stride 128}.
// Groups 2/3 zero (2-D tensor). Issued by one wave (wave-uniform branch: TDM
// ignores EXEC), completion via s_wait_tensorcnt 0 before the block barrier.

__device__ __forceinline__ void tdm_load_w_128x128(const bf16_t* __restrict__ WT,
                                                   bf16_t* __restrict__ sW) {
#if __has_builtin(__builtin_amdgcn_tensor_load_to_lds)
  unsigned lds_addr = (unsigned)(uintptr_t)(void*)sW;   // generic->LDS offset = addr[31:0]
  unsigned long long ga = (unsigned long long)(uintptr_t)WT;

  u32x4 g0;
  g0[0] = 1u;                                           // count=1 (valid), user mode
  g0[1] = lds_addr;                                     // lds_addr
  g0[2] = (unsigned)ga;                                 // global_addr[31:0]
  g0[3] = (unsigned)((ga >> 32) & 0x01FFFFFFu) | 0x80000000u; // addr[56:32] | type=2

  i32x8 g1;
  g1[0] = 1 << 16;                                      // data_size=1 (2 bytes)
  g1[1] = (int)(HDIM << 16);                            // tensor_dim0 lo16 = 128
  g1[2] = (int)(HDIM << 16);                            // dim0 hi=0 | tensor_dim1 lo16 = 128
  g1[3] = (int)(HDIM << 16);                            // dim1 hi=0 | tile_dim0 = 128
  g1[4] = HDIM;                                         // tile_dim1 = 128, tile_dim2 = 0
  g1[5] = HDIM;                                         // tensor_dim0_stride = 128
  g1[6] = 0;
  g1[7] = 0;

  i32x4 gz = {0, 0, 0, 0};

#if defined(__clang_major__) && __clang_major__ >= 23
  i32x8 gz8 = {0, 0, 0, 0, 0, 0, 0, 0};
  __builtin_amdgcn_tensor_load_to_lds(g0, g1, gz, gz, gz8, 0);
#else
  __builtin_amdgcn_tensor_load_to_lds(g0, g1, gz, gz, 0);
#endif
  __builtin_amdgcn_s_wait_tensorcnt(0);
#else
  // fallback: cooperative copy by the 32 lanes of wave 0 (the only caller)
  const uint4* wsv = (const uint4*)WT;
  uint4* wdv = (uint4*)sW;
  for (int i = (int)(threadIdx.x & 31); i < HDIM * HDIM / 8; i += 32) wdv[i] = wsv[i];
#endif
}

// ---------------- WMMA GEMM: out[N,128] = A_bf16[N,128] @ W[128,128] ----------------
// WT is W transposed: WT[n*128 + k]. Staged in LDS (32KB of 320KB) via TDM.
// Each wave computes a 16-row strip across all 8 column tiles; K loop = 4 WMMAs.

__global__ __launch_bounds__(256) void k_gemm128(const bf16_t* __restrict__ A,
                                                 const bf16_t* __restrict__ WT,
                                                 float* __restrict__ out, int nrows) {
  __shared__ bf16_t sW[HDIM * HDIM];
  int tid = threadIdx.x;
  int wave = tid >> 5, lane = tid & 31;
  if (wave == 0) {                       // wave-uniform: single TDM issue
    tdm_load_w_128x128(WT, sW);
  }
  __syncthreads();

  int m0 = (blockIdx.x * 8 + wave) * 16;
  if (m0 >= nrows) return;
  int lhalf = lane >> 4;          // 0: K-low half, 1: K-high half
  int lmod  = lane & 15;          // row (A) / column (B,C)

  // A fragments: 16x32 bf16 per k-step. lanes 0-15: K = k0+0..7,16..23;
  // lanes 16-31: K = k0+8..15,24..31  (two contiguous 16B chunks per lane).
  v16bf af[4];
  {
    const bf16_t* rp = A + (size_t)(m0 + lmod) * HDIM + lhalf * 8;
    #pragma unroll
    for (int kk = 0; kk < 4; ++kk) {
      union { uint4 u[2]; v16bf v; } t;
      t.u[0] = *(const uint4*)(rp + kk * 32);
      t.u[1] = *(const uint4*)(rp + kk * 32 + 16);
      af[kk] = t.v;
    }
  }

  #pragma unroll
  for (int nt = 0; nt < 8; ++nt) {
    int n0 = nt * 16;
    v8f acc = {};
    #pragma unroll
    for (int kk = 0; kk < 4; ++kk) {
      // B 32x16: column = lmod; lanes 0-15 hold K=k0+0..7,16..23, lanes 16-31 K=k0+8..15,24..31
      const bf16_t* cp = sW + (size_t)(n0 + lmod) * HDIM + kk * 32 + lhalf * 8;
      union { uint4 u[2]; v16bf v; } t;
      t.u[0] = *(const uint4*)cp;
      t.u[1] = *(const uint4*)(cp + 16);
      acc = __builtin_amdgcn_wmma_f32_16x16x32_bf16(false, af[kk], false, t.v,
                                                    (short)0, acc, false, false);
    }
    // C/D layout: VGPR r -> M = r + lhalf*8, N = lmod
    float* op = out + (size_t)(m0 + lhalf * 8) * HDIM + n0 + lmod;
    #pragma unroll
    for (int r = 0; r < 8; ++r) op[(size_t)r * HDIM] = acc[r];
  }
}

// ---------------- output head: d_out[N,16] = A_bf16[N,128] @ Wc[128,16] + bc ----------------

__global__ __launch_bounds__(256) void k_gemm_out(const bf16_t* __restrict__ A,
                                                  const bf16_t* __restrict__ WT,
                                                  const float* __restrict__ bc,
                                                  float* __restrict__ out, int nrows) {
  __shared__ bf16_t sW[CDIM * HDIM];   // 4KB, WT[n*128 + k], n in 0..15
  int tid = threadIdx.x;
  {
    const uint4* wsv = (const uint4*)WT;
    uint4* wdv = (uint4*)sW;
    if (tid < CDIM * HDIM / 8) wdv[tid] = wsv[tid];
  }
  __syncthreads();

  int wave = tid >> 5, lane = tid & 31;
  int m0 = (blockIdx.x * 8 + wave) * 16;
  if (m0 >= nrows) return;
  int lhalf = lane >> 4, lmod = lane & 15;

  v8f acc = {};
  const bf16_t* rp = A + (size_t)(m0 + lmod) * HDIM + lhalf * 8;
  #pragma unroll
  for (int kk = 0; kk < 4; ++kk) {
    union { uint4 u[2]; v16bf v; } ta, tb;
    ta.u[0] = *(const uint4*)(rp + kk * 32);
    ta.u[1] = *(const uint4*)(rp + kk * 32 + 16);
    const bf16_t* cp = sW + (size_t)lmod * HDIM + kk * 32 + lhalf * 8;
    tb.u[0] = *(const uint4*)cp;
    tb.u[1] = *(const uint4*)(cp + 16);
    acc = __builtin_amdgcn_wmma_f32_16x16x32_bf16(false, ta.v, false, tb.v,
                                                  (short)0, acc, false, false);
  }
  float bcv = bc[lmod];
  #pragma unroll
  for (int r = 0; r < 8; ++r)
    out[(size_t)(m0 + lhalf * 8 + r) * CDIM + lmod] = acc[r] + bcv;
}

// ---------------- GCN edge aggregation: agg[dst] += t[src] * norm ----------------
// wave per edge; lane 0 computes metadata, broadcast via shfl; float4 per lane.

__global__ __launch_bounds__(256) void k_edge_agg(const int* __restrict__ ei,
                                                  const float* __restrict__ ew,
                                                  const float* __restrict__ dinv,
                                                  const float* __restrict__ t,
                                                  float* __restrict__ agg, int E) {
  int lane = threadIdx.x & 31;
  long e = (long)blockIdx.x * 8 + (threadIdx.x >> 5);
  if (e >= E) return;
  int s = 0, dd = 0; float nrm = 0.f;
  if (lane == 0) {
    s  = ei[e];
    dd = ei[E + e];
    nrm = dinv[s] * ew[e] * dinv[dd];
  }
  s   = __shfl(s, 0, 32);
  dd  = __shfl(dd, 0, 32);
  nrm = __shfl(nrm, 0, 32);
  const float4 v = *(const float4*)(t + (size_t)s * HDIM + lane * 4);
  float* ap = agg + (size_t)dd * HDIM + lane * 4;
  atomicAdd(ap + 0, v.x * nrm);
  atomicAdd(ap + 1, v.y * nrm);
  atomicAdd(ap + 2, v.z * nrm);
  atomicAdd(ap + 3, v.w * nrm);
}

// fused self-loop (norm = dinv^2) + bias + ReLU, re-encode to bf16 for next GEMM
__global__ __launch_bounds__(256) void k_gcn_post(const float* __restrict__ agg,
                                                  const float* __restrict__ t,
                                                  const float* __restrict__ dinv,
                                                  const float* __restrict__ b,
                                                  bf16_t* __restrict__ hb, long total) {
  long i = (long)blockIdx.x * blockDim.x + threadIdx.x;
  if (i >= total) return;
  int node = (int)(i >> 7), c = (int)(i & 127);
  float di = dinv[node];
  float v = agg[i] + t[i] * di * di + b[c];
  hb[i] = (bf16_t)(v > 0.f ? v : 0.f);
}

// ---------------- GAT ----------------

__device__ __forceinline__ float warp_sum(float v) {
  #pragma unroll
  for (int off = 16; off > 0; off >>= 1) v += __shfl_down(v, off, 32);
  return v;
}

__device__ __forceinline__ float leaky(float a) { return a > 0.f ? a : 0.2f * a; }

// monotone float atomic-max via int/uint trick (works with -inf init)
__device__ __forceinline__ void atomicMaxF(float* addr, float v) {
  if (v >= 0.f) atomicMax((int*)addr, __float_as_int(v));
  else          atomicMin((unsigned int*)addr, __float_as_uint(v));
}

// wave per node: s[i] = g[i].att_src, d[i] = g[i].att_dst
__global__ __launch_bounds__(256) void k_dots(const float* __restrict__ g,
                                              const float* __restrict__ as_,
                                              const float* __restrict__ ad_,
                                              float* __restrict__ so,
                                              float* __restrict__ do_, int n) {
  int lane = threadIdx.x & 31;
  int node = blockIdx.x * 8 + (threadIdx.x >> 5);
  if (node >= n) return;
  const float4 gv = *(const float4*)(g + (size_t)node * HDIM + lane * 4);
  const float4 av = *(const float4*)(as_ + lane * 4);
  const float4 bv = *(const float4*)(ad_ + lane * 4);
  float ps = gv.x * av.x + gv.y * av.y + gv.z * av.z + gv.w * av.w;
  float pd = gv.x * bv.x + gv.y * bv.y + gv.z * bv.z + gv.w * bv.w;
  ps = warp_sum(ps);
  pd = warp_sum(pd);
  if (lane == 0) { so[node] = ps; do_[node] = pd; }
}

// thread per virtual edge (e < E real; e >= E self loop)
__global__ __launch_bounds__(256) void k_att_max(const int* __restrict__ ei,
                                                 const float* __restrict__ s,
                                                 const float* __restrict__ d,
                                                 float* __restrict__ amax, int E, int n) {
  long e = (long)blockIdx.x * blockDim.x + threadIdx.x;
  if (e >= (long)E + n) return;
  int si, di;
  if (e < E) { si = ei[e]; di = ei[E + e]; } else { si = di = (int)(e - E); }
  atomicMaxF(&amax[di], leaky(s[si] + d[di]));
}

__global__ __launch_bounds__(256) void k_att_den(const int* __restrict__ ei,
                                                 const float* __restrict__ s,
                                                 const float* __restrict__ d,
                                                 const float* __restrict__ amax,
                                                 float* __restrict__ den, int E, int n) {
  long e = (long)blockIdx.x * blockDim.x + threadIdx.x;
  if (e >= (long)E + n) return;
  int si, di;
  if (e < E) { si = ei[e]; di = ei[E + e]; } else { si = di = (int)(e - E); }
  float a = leaky(s[si] + d[di]);
  atomicAdd(&den[di], __expf(a - amax[di]));
}

// wave per virtual edge: agg[dst] += g[src] * alpha
__global__ __launch_bounds__(256) void k_att_agg(const int* __restrict__ ei,
                                                 const float* __restrict__ s,
                                                 const float* __restrict__ d,
                                                 const float* __restrict__ amax,
                                                 const float* __restrict__ den,
                                                 const float* __restrict__ g,
                                                 float* __restrict__ agg, int E, int n) {
  int lane = threadIdx.x & 31;
  long e = (long)blockIdx.x * 8 + (threadIdx.x >> 5);
  if (e >= (long)E + n) return;
  int si = 0, di = 0; float alpha = 0.f;
  if (lane == 0) {
    if (e < E) { si = ei[e]; di = ei[E + e]; } else { si = di = (int)(e - E); }
    float a = leaky(s[si] + d[di]);
    alpha = __expf(a - amax[di]) / den[di];
  }
  si = __shfl(si, 0, 32);
  di = __shfl(di, 0, 32);
  alpha = __shfl(alpha, 0, 32);
  const float4 v = *(const float4*)(g + (size_t)si * HDIM + lane * 4);
  float* ap = agg + (size_t)di * HDIM + lane * 4;
  atomicAdd(ap + 0, v.x * alpha);
  atomicAdd(ap + 1, v.y * alpha);
  atomicAdd(ap + 2, v.z * alpha);
  atomicAdd(ap + 3, v.w * alpha);
}

__global__ __launch_bounds__(256) void k_gat_post(const float* __restrict__ agg,
                                                  const float* __restrict__ bg,
                                                  bf16_t* __restrict__ hb, long total) {
  long i = (long)blockIdx.x * blockDim.x + threadIdx.x;
  if (i >= total) return;
  int c = (int)(i & 127);
  float v = agg[i] + bg[c];
  hb[i] = (bf16_t)(v > 0.f ? v : 0.f);
}

// ---------------------------------------------------------------------------

extern "C" void kernel_launch(void* const* d_in, const int* in_sizes, int n_in,
                              void* d_out, int out_size, void* d_ws, size_t ws_size,
                              hipStream_t stream) {
  const float* x    = (const float*)d_in[0];
  const int*   ei   = (const int*)d_in[1];
  const float* ew   = (const float*)d_in[2];
  const float* W1   = (const float*)d_in[3];
  const float* b1   = (const float*)d_in[4];
  const float* W2   = (const float*)d_in[5];
  const float* b2   = (const float*)d_in[6];
  const float* Wg   = (const float*)d_in[7];
  const float* asrc = (const float*)d_in[8];
  const float* adst = (const float*)d_in[9];
  const float* bg   = (const float*)d_in[10];
  const float* Wc   = (const float*)d_in[11];
  const float* bc   = (const float*)d_in[12];

  const int N = in_sizes[0] / HDIM;
  const int E = in_sizes[2];
  const long NF = (long)N * HDIM;

  // workspace carve-up (~131 MB total)
  char* w = (char*)d_ws;
  auto alignup = [](size_t v) { return (v + 255) & ~(size_t)255; };
  size_t off = 0;
  float*  tf   = (float*)(w + off);  off = alignup(off + (size_t)NF * 4);
  float*  agg  = (float*)(w + off);  off = alignup(off + (size_t)NF * 4);
  bf16_t* hbf  = (bf16_t*)(w + off); off = alignup(off + (size_t)NF * 2);
  bf16_t* w1t  = (bf16_t*)(w + off); off = alignup(off + (size_t)HDIM * HDIM * 2);
  bf16_t* w2t  = (bf16_t*)(w + off); off = alignup(off + (size_t)HDIM * HDIM * 2);
  bf16_t* wgt  = (bf16_t*)(w + off); off = alignup(off + (size_t)HDIM * HDIM * 2);
  bf16_t* wct  = (bf16_t*)(w + off); off = alignup(off + (size_t)HDIM * CDIM * 2);
  float*  deg  = (float*)(w + off);  off = alignup(off + (size_t)N * 4);
  float*  dinv = (float*)(w + off);  off = alignup(off + (size_t)N * 4);
  float*  sarr = (float*)(w + off);  off = alignup(off + (size_t)N * 4);
  float*  darr = (float*)(w + off);  off = alignup(off + (size_t)N * 4);
  float*  amax = (float*)(w + off);  off = alignup(off + (size_t)N * 4);
  float*  den  = (float*)(w + off);  off = alignup(off + (size_t)N * 4);

  dim3 blk(256);
  auto g1 = [](long n) { return dim3((unsigned)((n + 255) / 256)); };    // thread/item
  auto gw = [](long n) { return dim3((unsigned)((n + 7) / 8)); };        // wave/item
  unsigned gemm_blocks = (unsigned)(((N + 15) / 16 + 7) / 8);

  // degrees (self loop weight 1 folded into init)
  k_fill_f32<<<g1(N), blk, 0, stream>>>(deg, N, 1.0f);
  k_deg<<<g1(E), blk, 0, stream>>>(ei, ew, deg, E);
  k_dinv<<<g1(N), blk, 0, stream>>>(deg, dinv, N);

  // weights -> transposed bf16
  k_wT_to_bf<<<g1(HDIM * HDIM), blk, 0, stream>>>(W1, w1t, HDIM, HDIM);
  k_wT_to_bf<<<g1(HDIM * HDIM), blk, 0, stream>>>(W2, w2t, HDIM, HDIM);
  k_wT_to_bf<<<g1(HDIM * HDIM), blk, 0, stream>>>(Wg, wgt, HDIM, HDIM);
  k_wT_to_bf<<<g1(HDIM * CDIM), blk, 0, stream>>>(Wc, wct, HDIM, CDIM);

  // x -> bf16
  k_f32_to_bf<<<g1(NF), blk, 0, stream>>>(x, hbf, NF);

  // ---- GCN layer 1 ----
  k_gemm128<<<gemm_blocks, blk, 0, stream>>>(hbf, w1t, tf, N);
  k_fill_f32<<<g1(NF), blk, 0, stream>>>(agg, NF, 0.f);
  k_edge_agg<<<gw(E), blk, 0, stream>>>(ei, ew, dinv, tf, agg, E);
  k_gcn_post<<<g1(NF), blk, 0, stream>>>(agg, tf, dinv, b1, hbf, NF);

  // ---- GCN layer 2 ----
  k_gemm128<<<gemm_blocks, blk, 0, stream>>>(hbf, w2t, tf, N);
  k_fill_f32<<<g1(NF), blk, 0, stream>>>(agg, NF, 0.f);
  k_edge_agg<<<gw(E), blk, 0, stream>>>(ei, ew, dinv, tf, agg, E);
  k_gcn_post<<<g1(NF), blk, 0, stream>>>(agg, tf, dinv, b2, hbf, NF);

  // ---- GAT ----
  k_gemm128<<<gemm_blocks, blk, 0, stream>>>(hbf, wgt, tf, N);   // g in tf
  k_dots<<<gw(N), blk, 0, stream>>>(tf, asrc, adst, sarr, darr, N);
  k_fill_f32<<<g1(N), blk, 0, stream>>>(amax, N, -INFINITY);
  k_fill_f32<<<g1(N), blk, 0, stream>>>(den, N, 0.f);
  long EV = (long)E + N;
  k_att_max<<<g1(EV), blk, 0, stream>>>(ei, sarr, darr, amax, E, N);
  k_att_den<<<g1(EV), blk, 0, stream>>>(ei, sarr, darr, amax, den, E, N);
  k_fill_f32<<<g1(NF), blk, 0, stream>>>(agg, NF, 0.f);
  k_att_agg<<<gw(EV), blk, 0, stream>>>(ei, sarr, darr, amax, den, tf, agg, E, N);
  k_gat_post<<<g1(NF), blk, 0, stream>>>(agg, bg, hbf, NF);

  // ---- output head ----
  k_gemm_out<<<gemm_blocks, blk, 0, stream>>>(hbf, wct, bc, (float*)d_out, N);
}
